// AttentionBlock_8100308320352
// MI455X (gfx1250) — matrix operand order repaired and checked by reading, not compiled, not run on hardware
//
#include <hip/hip_runtime.h>
#include <hip/hip_bf16.h>

#define D_MODEL 1024
#define N_HEADS 16
#define D_HEAD  64
#define SEQ_T   2048
#define BATCH   2
#define ROWS    (BATCH * SEQ_T)   // 4096
#define D3      (3 * D_MODEL)     // 3072

typedef __bf16 bf16_t;
typedef __attribute__((ext_vector_type(16))) __bf16 v16bf;
typedef __attribute__((ext_vector_type(8)))  __bf16 v8bf;
typedef __attribute__((ext_vector_type(8)))  float  v8f;

static __device__ __forceinline__ v8f zero8() {
  v8f z;
#pragma unroll
  for (int i = 0; i < 8; ++i) z[i] = 0.0f;
  return z;
}

// A fragment: 16x32 bf16 tile of row-major A (row stride ld), tile origin (m0,k0).
// Lane L: row = m0 + (L&15); halves a[0..7] = K k0+kb.., a[8..15] = K k0+16+kb..,
// kb = 8*(L>>4).  (ISA 7.12.2, 16-bit A 16x32 layout)
static __device__ __forceinline__ v16bf load_a_frag(const bf16_t* A, int ld, int m0,
                                                    int k0, int lane) {
  int row = m0 + (lane & 15);
  int kb  = (lane >> 4) << 3;
  const bf16_t* p = A + (size_t)row * ld + k0 + kb;
  v8bf lo = *(const v8bf*)(p);
  v8bf hi = *(const v8bf*)(p + 16);
  v16bf a;
#pragma unroll
  for (int i = 0; i < 8; ++i) { a[i] = lo[i]; a[i + 8] = hi[i]; }
  return a;
}

// B fragment: 32x16 bf16 tile, B supplied TRANSPOSED row-major (Bt[N][K], stride ld).
// Lane L: col N = n0 + (L&15); b[0..7] = K k0+kb.., b[8..15] = K k0+16+kb.., kb = 8*(L>>4).
static __device__ __forceinline__ v16bf load_b_frag(const bf16_t* Bt, int ld, int n0,
                                                    int k0, int lane) {
  const bf16_t* p = Bt + (size_t)(n0 + (lane & 15)) * ld + k0 + ((lane >> 4) << 3);
  v8bf lo = *(const v8bf*)(p);
  v8bf hi = *(const v8bf*)(p + 16);
  v16bf b;
#pragma unroll
  for (int i = 0; i < 8; ++i) { b[i] = lo[i]; b[i + 8] = hi[i]; }
  return b;
}

static __device__ __forceinline__ v8f wmma_bf16(v16bf a, v16bf b, v8f c) {
  return __builtin_amdgcn_wmma_f32_16x16x32_bf16(false, a, false, b, (short)0, c,
                                                 false, false);
}

// ---------------------------------------------------------------- weights -> bf16^T
__global__ void k_convert_w(const float* __restrict__ qkv_w,
                            const float* __restrict__ out_w,
                            bf16_t* __restrict__ qkv_wt,
                            bf16_t* __restrict__ out_wt) {
  size_t i  = (size_t)blockIdx.x * blockDim.x + threadIdx.x;
  size_t n1 = (size_t)D_MODEL * D3;
  if (i < n1) {
    size_t k = i / D3, n = i % D3;                  // qkv_w[k][n]
    qkv_wt[n * D_MODEL + k] = (bf16_t)qkv_w[i];
  } else {
    size_t j = i - n1;
    if (j < (size_t)D_MODEL * D_MODEL) {
      size_t k = j / D_MODEL, n = j % D_MODEL;
      out_wt[n * D_MODEL + k] = (bf16_t)out_w[j];
    }
  }
}

// ---------------------------------------------------------------- layernorm -> bf16
__global__ void __launch_bounds__(256) k_layernorm(const float* __restrict__ x,
                                                   const float* __restrict__ g,
                                                   const float* __restrict__ bt,
                                                   bf16_t* __restrict__ h) {
  int row = blockIdx.x;                              // 0..4095
  int tid = threadIdx.x;                             // 256 threads, 4 elems each
  const float* xr = x + (size_t)row * D_MODEL;
  float4 v = ((const float4*)xr)[tid];
  float e[4] = { v.x, v.y, v.z, v.w };
  float s = 0.f, ss = 0.f;
#pragma unroll
  for (int j = 0; j < 4; ++j) { s += e[j]; ss += e[j] * e[j]; }
  __shared__ float rs[256], rss[256];
  rs[tid] = s; rss[tid] = ss;
  __syncthreads();
  for (int o = 128; o > 0; o >>= 1) {
    if (tid < o) { rs[tid] += rs[tid + o]; rss[tid] += rss[tid + o]; }
    __syncthreads();
  }
  float mu  = rs[0] * (1.0f / D_MODEL);
  float var = rss[0] * (1.0f / D_MODEL) - mu * mu;
  float inv = rsqrtf(var + 1e-5f);
  int c = tid * 4;
#pragma unroll
  for (int j = 0; j < 4; ++j)
    h[(size_t)row * D_MODEL + c + j] = (bf16_t)((e[j] - mu) * inv * g[c + j] + bt[c + j]);
}

// ---------------------------------------------------------------- QKV GEMM (WMMA)
// Register-blocked: one wave computes a 32(M) x 64(N) tile => 8 WMMAs per K-step
// from 2 A-frags + 4 B-frags (0.75 frag loads per WMMA, B reused 2x).
// qkv = h @ W + b  -> scatter into Q[t,d], K[t,d], V^T[d,t] per (batch,head), bf16.
__global__ void __launch_bounds__(256) k_qkv_gemm(const bf16_t* __restrict__ h,
                                                  const bf16_t* __restrict__ wt,
                                                  const float* __restrict__ bias,
                                                  bf16_t* __restrict__ qb,
                                                  bf16_t* __restrict__ kb,
                                                  bf16_t* __restrict__ vt) {
  int lane = threadIdx.x & 31;
  int w    = blockIdx.x * (blockDim.x >> 5) + (threadIdx.x >> 5);
  const int NT = D3 / 64;                            // 48 N-blocks
  int mt = w / NT, nt = w % NT;
  int m0 = mt * 32, n0 = nt * 64;

  v8f acc[2][4];
#pragma unroll
  for (int r = 0; r < 2; ++r)
#pragma unroll
    for (int s = 0; s < 4; ++s) acc[r][s] = zero8();

  for (int k = 0; k < D_MODEL; k += 32) {
    if (k + 128 < D_MODEL) {                         // stream weights ahead
      __builtin_prefetch(wt + (size_t)(n0 + (lane & 15)) * D_MODEL + k + 128, 0, 3);
    }
    v16bf a0 = load_a_frag(h, D_MODEL, m0,      k, lane);
    v16bf a1 = load_a_frag(h, D_MODEL, m0 + 16, k, lane);
#pragma unroll
    for (int s = 0; s < 4; ++s) {
      v16bf b = load_b_frag(wt, D_MODEL, n0 + s * 16, k, lane);
      acc[0][s] = wmma_bf16(a0, b, acc[0][s]);
      acc[1][s] = wmma_bf16(a1, b, acc[1][s]);
    }
  }

  int n = lane & 15, hi = lane >> 4;
  int which = n0 >> 10;                              // constant per wave (64 | 1024)
  int head  = (n0 & (D_MODEL - 1)) >> 6;
#pragma unroll
  for (int s = 0; s < 4; ++s) {
    int col  = n0 + s * 16 + n;
    int d    = col & (D_HEAD - 1);
    float bv = bias[col];
#pragma unroll
    for (int r = 0; r < 2; ++r) {
#pragma unroll
      for (int i = 0; i < 8; ++i) {
        int rowg = m0 + r * 16 + i + 8 * hi;         // global token row 0..4095
        int bidx = rowg >> 11;
        int t    = rowg & (SEQ_T - 1);
        float val = acc[r][s][i] + bv;
        size_t bh = (size_t)(bidx * N_HEADS + head);
        if (which == 0)      qb[(bh * SEQ_T + t) * D_HEAD + d] = (bf16_t)val;
        else if (which == 1) kb[(bh * SEQ_T + t) * D_HEAD + d] = (bf16_t)val;
        else                 vt[(bh * D_HEAD + d) * SEQ_T + t] = (bf16_t)val;
      }
    }
  }
}

// ---------------------------------------------------------------- flash attention
// One wave per (b,h,16-query tile). Pass1: online softmax stats via WMMA scores.
// Pass2: recompute scores, write normalized probs (fp32 to d_out + bf16 to LDS),
// and accumulate O = P @ V via WMMA (probs re-layout C->A through 1KB LDS tile).
__global__ void __launch_bounds__(32) k_attention(const bf16_t* __restrict__ qb,
                                                  const bf16_t* __restrict__ kbuf,
                                                  const bf16_t* __restrict__ vt,
                                                  float* __restrict__ attn,
                                                  bf16_t* __restrict__ ob) {
  __shared__ bf16_t sP[16 * 32];
  int lane = threadIdx.x;
  int qt = blockIdx.x & (SEQ_T / 16 - 1);            // 0..127
  int bh = blockIdx.x >> 7;                           // 0..31
  int b  = bh >> 4, hd = bh & (N_HEADS - 1);
  int q0 = qt * 16;
  const bf16_t* Q = qb   + (size_t)bh * SEQ_T * D_HEAD;
  const bf16_t* K = kbuf + (size_t)bh * SEQ_T * D_HEAD;
  const bf16_t* V = vt   + (size_t)bh * D_HEAD * SEQ_T;
  float* A = attn + (size_t)bh * SEQ_T * SEQ_T;

  v16bf aQ0 = load_a_frag(Q, D_HEAD, q0, 0, lane);
  v16bf aQ1 = load_a_frag(Q, D_HEAD, q0, 32, lane);
  int n = lane & 15, hi = lane >> 4;
  int nk = qt + 1;                                   // causal key tiles

  float mrun[8], lrun[8];
#pragma unroll
  for (int i = 0; i < 8; ++i) { mrun[i] = -3.0e38f; lrun[i] = 0.f; }

  // ---- pass 1: row max / row sum (online)
  for (int j = 0; j < nk; ++j) {
    v8f c = zero8();
    c = wmma_bf16(aQ0, load_b_frag(K, D_HEAD, j * 16, 0, lane), c);
    c = wmma_bf16(aQ1, load_b_frag(K, D_HEAD, j * 16, 32, lane), c);
    int kcol = j * 16 + n;
#pragma unroll
    for (int i = 0; i < 8; ++i) {
      float s = c[i] * 0.125f;                       // 1/sqrt(64)
      int qrow = q0 + i + 8 * hi;
      if (kcol > qrow) s = -3.0e38f;
      float v = s;
      for (int o = 1; o < 16; o <<= 1) v = fmaxf(v, __shfl_xor(v, o, 32));
      float mnew = fmaxf(mrun[i], v);
      float p = __expf(s - mnew);
      float ps = p;
      for (int o = 1; o < 16; o <<= 1) ps += __shfl_xor(ps, o, 32);
      lrun[i] = lrun[i] * __expf(mrun[i] - mnew) + ps;
      mrun[i] = mnew;
    }
  }
  float rcl[8];
#pragma unroll
  for (int i = 0; i < 8; ++i) rcl[i] = 1.0f / lrun[i];

  // ---- pass 2: normalized probs out + O = P @ V
  v8f oacc[4];
#pragma unroll
  for (int t2 = 0; t2 < 4; ++t2) oacc[t2] = zero8();

  int nj2 = (nk + 1) >> 1;
  for (int j2 = 0; j2 < nj2; ++j2) {
#pragma unroll
    for (int sub = 0; sub < 2; ++sub) {
      int j = j2 * 2 + sub;
      if (j < nk) {
        v8f c = zero8();
        c = wmma_bf16(aQ0, load_b_frag(K, D_HEAD, j * 16, 0, lane), c);
        c = wmma_bf16(aQ1, load_b_frag(K, D_HEAD, j * 16, 32, lane), c);
        int kcol = j * 16 + n;
#pragma unroll
        for (int i = 0; i < 8; ++i) {
          float s = c[i] * 0.125f;
          int mi = i + 8 * hi;
          int qrow = q0 + mi;
          float p = (kcol > qrow) ? 0.f : __expf(s - mrun[i]) * rcl[i];
          A[(size_t)qrow * SEQ_T + kcol] = p;
          sP[mi * 32 + sub * 16 + n] = (bf16_t)p;
        }
      } else {
#pragma unroll
        for (int i = 0; i < 8; ++i) sP[(i + 8 * hi) * 32 + sub * 16 + n] = (bf16_t)0.f;
      }
    }
    __syncthreads();
    v16bf aP = load_a_frag(sP, 32, 0, 0, lane);      // C->A re-layout via LDS
#pragma unroll
    for (int t2 = 0; t2 < 4; ++t2) {
      v16bf bv = load_b_frag(V, SEQ_T, t2 * 16, j2 * 32, lane);
      oacc[t2] = wmma_bf16(aP, bv, oacc[t2]);
    }
    __syncthreads();
  }

  // O tile -> ob[token, h*64 + d] (bf16)
#pragma unroll
  for (int t2 = 0; t2 < 4; ++t2) {
    int col = hd * D_HEAD + t2 * 16 + n;
#pragma unroll
    for (int i = 0; i < 8; ++i) {
      int rowg = b * SEQ_T + q0 + i + 8 * hi;
      ob[(size_t)rowg * D_MODEL + col] = (bf16_t)oacc[t2][i];
    }
  }

  // zero the causal upper triangle (harness poisons d_out)
  int c0 = nk * 16;
  for (int r = 0; r < 16; ++r) {
    float* Ar = A + (size_t)(q0 + r) * SEQ_T;
    for (int cidx = c0 + lane; cidx < SEQ_T; cidx += 32) Ar[cidx] = 0.f;
  }
}

// ---------------------------------------------------------------- out proj (WMMA)
// Register-blocked 32x64 per wave, same scheme as k_qkv_gemm.
__global__ void __launch_bounds__(256) k_out_proj(const bf16_t* __restrict__ ob,
                                                  const bf16_t* __restrict__ wt,
                                                  const float* __restrict__ bias,
                                                  const float* __restrict__ x,
                                                  float* __restrict__ xout) {
  int lane = threadIdx.x & 31;
  int w    = blockIdx.x * (blockDim.x >> 5) + (threadIdx.x >> 5);
  const int NT = D_MODEL / 64;                       // 16 N-blocks
  int mt = w / NT, nt = w % NT;
  int m0 = mt * 32, n0 = nt * 64;

  v8f acc[2][4];
#pragma unroll
  for (int r = 0; r < 2; ++r)
#pragma unroll
    for (int s = 0; s < 4; ++s) acc[r][s] = zero8();

  for (int k = 0; k < D_MODEL; k += 32) {
    if (k + 128 < D_MODEL) {
      __builtin_prefetch(wt + (size_t)(n0 + (lane & 15)) * D_MODEL + k + 128, 0, 3);
    }
    v16bf a0 = load_a_frag(ob, D_MODEL, m0,      k, lane);
    v16bf a1 = load_a_frag(ob, D_MODEL, m0 + 16, k, lane);
#pragma unroll
    for (int s = 0; s < 4; ++s) {
      v16bf b = load_b_frag(wt, D_MODEL, n0 + s * 16, k, lane);
      acc[0][s] = wmma_bf16(a0, b, acc[0][s]);
      acc[1][s] = wmma_bf16(a1, b, acc[1][s]);
    }
  }

  int n = lane & 15, hi = lane >> 4;
#pragma unroll
  for (int s = 0; s < 4; ++s) {
    int col  = n0 + s * 16 + n;
    float bv = bias[col];
#pragma unroll
    for (int r = 0; r < 2; ++r) {
#pragma unroll
      for (int i = 0; i < 8; ++i) {
        int row = m0 + r * 16 + i + 8 * hi;
        size_t idx = (size_t)row * D_MODEL + col;
        xout[idx] = x[idx] + acc[r][s][i] + bv;
      }
    }
  }
}

// ----------------------------------------------------------------------------
extern "C" void kernel_launch(void* const* d_in, const int* in_sizes, int n_in,
                              void* d_out, int out_size, void* d_ws, size_t ws_size,
                              hipStream_t stream) {
  const float* x     = (const float*)d_in[0];
  const float* ln_g  = (const float*)d_in[1];
  const float* ln_b  = (const float*)d_in[2];
  const float* qkv_w = (const float*)d_in[3];
  const float* qkv_b = (const float*)d_in[4];
  const float* out_w = (const float*)d_in[5];
  const float* out_b = (const float*)d_in[6];

  float* xout = (float*)d_out;                                   // [B,T,D]
  float* attn = xout + (size_t)ROWS * D_MODEL;                   // [B,H,T,T]

  char* ws = (char*)d_ws;
  bf16_t* h_bf   = (bf16_t*)ws;  ws += (size_t)ROWS * D_MODEL * 2;          // 8 MB
  bf16_t* qkv_wt = (bf16_t*)ws;  ws += (size_t)D_MODEL * D3 * 2;            // 6 MB
  bf16_t* out_wt = (bf16_t*)ws;  ws += (size_t)D_MODEL * D_MODEL * 2;       // 2 MB
  bf16_t* q_buf  = (bf16_t*)ws;  ws += (size_t)ROWS * D_MODEL * 2;          // 8 MB
  bf16_t* k_buf  = (bf16_t*)ws;  ws += (size_t)ROWS * D_MODEL * 2;          // 8 MB
  bf16_t* vt_buf = (bf16_t*)ws;  ws += (size_t)ROWS * D_MODEL * 2;          // 8 MB
  bf16_t* o_buf  = (bf16_t*)ws;  ws += (size_t)ROWS * D_MODEL * 2;          // 8 MB

  {
    size_t tot = (size_t)D_MODEL * D3 + (size_t)D_MODEL * D_MODEL;
    int blocks = (int)((tot + 255) / 256);
    k_convert_w<<<blocks, 256, 0, stream>>>(qkv_w, out_w, qkv_wt, out_wt);
  }
  k_layernorm<<<ROWS, 256, 0, stream>>>(x, ln_g, ln_b, h_bf);
  {
    int waves = (ROWS / 32) * (D3 / 64);             // 128 * 48 = 6144
    k_qkv_gemm<<<waves / 8, 256, 0, stream>>>(h_bf, qkv_wt, qkv_b,
                                              q_buf, k_buf, vt_buf);
  }
  k_attention<<<BATCH * N_HEADS * (SEQ_T / 16), 32, 0, stream>>>(
      q_buf, k_buf, vt_buf, attn, o_buf);
  {
    int waves = (ROWS / 32) * (D_MODEL / 64);        // 128 * 16 = 2048
    k_out_proj<<<waves / 8, 256, 0, stream>>>(o_buf, out_wt, out_b, x, xout);
  }
}
